// Transformer_25933012533684
// MI455X (gfx1250) — compile-verified, matrix operand order checked by reading, not run on hardware
//
#include <hip/hip_runtime.h>
#include <hip/hip_bf16.h>
#include <math.h>

// ---------------- types ----------------
typedef __attribute__((ext_vector_type(16))) __bf16 v16bf;
typedef __attribute__((ext_vector_type(8)))  float  v8f;
typedef __attribute__((ext_vector_type(4)))  unsigned v4u;
typedef __attribute__((ext_vector_type(8)))  int    v8i;
typedef __attribute__((ext_vector_type(4)))  int    v4i;
typedef unsigned short u16;
typedef unsigned int   u32;
typedef unsigned long long u64;

#define B_   2
#define S_   2048
#define D_   1024
#define H_   16
#define HD_  64
#define NT_  (B_ * S_)    /* 4096 token rows */
#define DFF_ (4 * D_)     /* 4096 */

union FragB { uint4 u[2]; v16bf v; };   // 32B = 16 bf16 (one WMMA A/B operand)
union AccF  { v8f  v;  float f[8]; };   // 16x16 f32 C/D tile slice per lane

__device__ __forceinline__ u16 f2bf(float f) {
  union { float f; u32 u; } x; x.f = f;
  u32 r = x.u + 0x7FFFu + ((x.u >> 16) & 1u);  // round-to-nearest-even
  return (u16)(r >> 16);
}
__device__ __forceinline__ float gelu_tanh(float x) {
  const float c = 0.7978845608028654f; // sqrt(2/pi)
  return 0.5f * x * (1.0f + tanhf(c * (x + 0.044715f * x * x * x)));
}
__device__ __forceinline__ v8f wmma_bf16(const v16bf& a, const v16bf& b, const v8f& c) {
  return __builtin_amdgcn_wmma_f32_16x16x32_bf16(false, a, false, b, (short)0, c, false, false);
}

// LDS byte offset of a __shared__ object (generic-pointer low 32 bits == DS offset)
__device__ __forceinline__ u32 lds_off(const void* p) {
  return (u32)(u64)p;
}
// CDNA5 async global->LDS mover: 16B per lane, tracked by ASYNCcnt.
__device__ __forceinline__ void async_ld_b128(u32 lds_byte, u32 gl_byte_off, const void* base) {
  asm volatile("global_load_async_to_lds_b128 %0, %1, %2"
               :: "v"(lds_byte), "v"(gl_byte_off), "s"((u64)base)
               : "memory");
}
__device__ __forceinline__ void wait_async0() {
  asm volatile("s_wait_asynccnt 0x0" ::: "memory");
}

// CDNA5 Tensor Data Mover: DMA a 2D tile (tile_d1 rows x tile_d0 elems, 2B/elem)
// from global (row stride stride_d0 elems) into LDS (contiguous), TENSORcnt-tracked.
// D# bit layout per CDNA5 ISA ch.8 (group0 128b, group1 256b; groups 2/3 unused->0).
__device__ __forceinline__ void tdm_load_2d(u32 lds_byte, u64 gaddr,
                                            u32 tile_d0, u32 tile_d1,
                                            u32 tens_d0, u32 tens_d1, u32 stride_d0) {
  v4u g0;
  g0[0] = 1u;                                 // count=1 (valid user descriptor)
  g0[1] = lds_byte;                           // lds_addr [63:32]
  g0[2] = (u32)gaddr;                         // global_addr [95:64]
  g0[3] = (u32)((gaddr >> 32) & 0x01FFFFFFu)  // global_addr [120:96]
          | 0x80000000u;                      // type=2 ("image") [127:126]
  v8i g1;
  g1[0] = (int)(1u << 16);                    // data_size=1 -> 2 bytes; no multicast
  g1[1] = (int)((tens_d0 & 0xFFFFu) << 16);   // tensor_dim0[15:0] @ bits 63:48
  g1[2] = (int)((tens_d0 >> 16) | ((tens_d1 & 0xFFFFu) << 16)); // dim0 hi | dim1 lo
  g1[3] = (int)((tens_d1 >> 16) | (tile_d0 << 16));             // dim1 hi | tile_dim0
  g1[4] = (int)(tile_d1 & 0xFFFFu);           // tile_dim1 (tile_dim2=0)
  g1[5] = (int)stride_d0;                     // tensor_dim0_stride[31:0]
  g1[6] = 0;                                  // stride hi | dim1_stride lo (2D)
  g1[7] = 0;
  v4i z4 = {0, 0, 0, 0};
#if defined(__clang_major__) && (__clang_major__ >= 23)
  v8i z8 = {0, 0, 0, 0, 0, 0, 0, 0};
  __builtin_amdgcn_tensor_load_to_lds(g0, g1, z4, z4, z8, 0);
#else
  __builtin_amdgcn_tensor_load_to_lds(g0, g1, z4, z4, 0);
#endif
}

// ---------------- elementwise prep kernels ----------------
__global__ void k_cast_bf16(const float* __restrict__ in, u16* __restrict__ out, int n) {
  int i = blockIdx.x * blockDim.x + threadIdx.x;
  if (i < n) out[i] = f2bf(in[i]);
}

// W [K,N] f32 -> Wt [N,K] bf16 (so GEMM B-operand rows are contiguous along K)
__global__ void k_transpose_w(const float* __restrict__ w, u16* __restrict__ wt, int K, int N) {
  int i = blockIdx.x * blockDim.x + threadIdx.x;
  if (i < K * N) { int k = i / N, n = i - k * N; wt[(size_t)n * K + k] = f2bf(w[i]); }
}

// V bf16 [B,S,H,HD] -> Vt bf16 [B,H,HD,S]  (PV GEMM B-operand = V^T, contiguous in S)
__global__ void k_transpose_v(const u16* __restrict__ v, u16* __restrict__ vt) {
  int i = blockIdx.x * blockDim.x + threadIdx.x;
  if (i < NT_ * D_) {
    int row = i / D_, col = i - row * D_;
    int b = row / S_, s = row - b * S_;
    int h = col / HD_, hd = col - h * HD_;
    vt[(((size_t)(b * H_ + h) * HD_) + hd) * S_ + s] = v[i];
  }
}

// ---------------- tiled WMMA GEMM (TDM + async, double-buffered LDS) ----------------
// C[M,N] = epilogue(A_bf16[M,K] @ Bt_bf16[N,K]^T + bias)
// block: 256 thr = 8 waves; block tile 128(M) x 128(N); K-step 32.
// wave tile 32(M) x 64(N) -> 2x4 = 8 WMMA per K-step.
// A tile staged via per-thread global_load_async_to_lds_b128 (ASYNCcnt);
// B (weight) tile staged via one tensor_load_to_lds DMA issued by wave 0 (TENSORcnt).
#define EP_BF16 0   // outb = val
#define EP_RES  1   // outf = res + val
#define EP_GELU 2   // outb = gelu(val)

template <int EP>
__global__ __launch_bounds__(256)
void k_gemm(const u16* __restrict__ A, const u16* __restrict__ Bt,
            const float* __restrict__ bias, int M, int N, int K,
            const float* __restrict__ res, float* __restrict__ outf,
            u16* __restrict__ outb) {
  __shared__ u16 sA[2][128 * 32];
  __shared__ u16 sB[2][128 * 32];

  const int tid  = threadIdx.x;
  const int lane = tid & 31;
  const int w    = tid >> 5;
  const int ln   = lane & 15;
  const int half = lane >> 4;
  const int m0   = blockIdx.y * 128;
  const int n0   = blockIdx.x * 128;
  const int wm   = (w & 3) * 32;   // wave M offset in block tile
  const int wn   = (w >> 2) * 64;  // wave N offset in block tile

  // A staging map: 128x32 bf16 tile = 512 x uint4; thread covers vectors tid, tid+256
  const int sr = tid >> 2;          // rows 0..63 (and +64)
  const int sc = (tid & 3) * 8;     // bf16 col 0/8/16/24

  v8f acc[2][4];
  const v8f z8 = {0.f,0.f,0.f,0.f,0.f,0.f,0.f,0.f};
  for (int mi = 0; mi < 2; ++mi)
    for (int ni = 0; ni < 4; ++ni) acc[mi][ni] = z8;

  auto stage = [&](int buf, int k0) {
    // A tile rows m0..m0+127 (async mover, 2 wave-instructions per wave)
    async_ld_b128(lds_off(&sA[buf][sr * 32 + sc]),
                  (u32)(((sr + m0) * K + k0 + sc) * 2), A);
    async_ld_b128(lds_off(&sA[buf][(sr + 64) * 32 + sc]),
                  (u32)(((sr + 64 + m0) * K + k0 + sc) * 2), A);
    // B tile rows n0..n0+127 (one TDM descriptor, wave 0 only; EXEC-independent)
    if (w == 0)
      tdm_load_2d(lds_off(&sB[buf][0]),
                  (u64)Bt + ((size_t)n0 * K + k0) * 2,
                  /*tile*/32, 128, /*tensor*/(u32)K, (u32)N, /*stride*/(u32)K);
  };

  stage(0, 0);
  int buf = 0;
  for (int k0 = 0; k0 < K; k0 += 32, buf ^= 1) {
    wait_async0();                                  // my A-tile async fills landed
    if (w == 0) __builtin_amdgcn_s_wait_tensorcnt(0); // B-tile DMA landed
    __syncthreads();   // everyone's fills visible; everyone done reading buf^1
    if (k0 + 32 < K) stage(buf ^ 1, k0 + 32);       // overlap next fill with compute

    // fragments from LDS (A-layout: lane = row, K 0..7|8..15 by lane half, +16)
    FragB fa[2], fb[4];
    for (int mi = 0; mi < 2; ++mi) {
      int r = wm + mi * 16 + ln;
      fa[mi].u[0] = *(const uint4*)(&sA[buf][r * 32 + half * 8]);
      fa[mi].u[1] = *(const uint4*)(&sA[buf][r * 32 + 16 + half * 8]);
    }
    for (int ni = 0; ni < 4; ++ni) {
      int r = wn + ni * 16 + ln;
      fb[ni].u[0] = *(const uint4*)(&sB[buf][r * 32 + half * 8]);
      fb[ni].u[1] = *(const uint4*)(&sB[buf][r * 32 + 16 + half * 8]);
    }
    for (int mi = 0; mi < 2; ++mi)
      for (int ni = 0; ni < 4; ++ni)
        acc[mi][ni] = wmma_bf16(fa[mi].v, fb[ni].v, acc[mi][ni]);
  }

  // epilogue: C-layout = lane col n, VGPR v -> row (v + 8*half)
  for (int mi = 0; mi < 2; ++mi)
    for (int ni = 0; ni < 4; ++ni) {
      AccF a; a.v = acc[mi][ni];
      int n = n0 + wn + ni * 16 + ln;
      float bn = bias[n];
      for (int vv = 0; vv < 8; ++vv) {
        int m = m0 + wm + mi * 16 + half * 8 + vv;
        float val = a.f[vv] + bn;
        size_t idx = (size_t)m * N + n;
        if (EP == EP_BF16)      outb[idx] = f2bf(val);
        else if (EP == EP_RES)  outf[idx] = res[idx] + val;
        else                    outb[idx] = f2bf(gelu_tanh(val));
      }
    }
}

// ---------------- flash attention (causal, online softmax) ----------------
// block = 128 thr (4 waves); block handles (b,h) x 64 q-rows; wave owns 16 q-rows.
// Q,K bf16 [B,S,H,HD]; Vt bf16 [B,H,HD,S]; Y bf16 [B*S, D]
__global__ __launch_bounds__(128)
void k_flash(const u16* __restrict__ Q, const u16* __restrict__ Km,
             const u16* __restrict__ Vt, u16* __restrict__ Y) {
  __shared__ u16 sP[4][16 * 32];  // per-wave P tile (C-layout -> A-layout via LDS)

  const int tid  = threadIdx.x;
  const int lane = tid & 31;
  const int w    = tid >> 5;
  const int ln   = lane & 15;
  const int half = lane >> 4;
  const int bh   = blockIdx.y;
  const int b    = bh / H_, h = bh - b * H_;
  const int qb   = blockIdx.x * 64 + w * 16;  // wave's first query row
  const float scale = 0.125f;                 // 1/sqrt(64)
  const v8f z8 = {0.f,0.f,0.f,0.f,0.f,0.f,0.f,0.f};

  // Q fragments (K=64 -> two 16x32 steps), rows strided by D in memory
  FragB fq[2];
  {
    const u16* qrow = Q + (size_t)(b * S_ + qb + ln) * D_ + h * HD_;
    for (int s = 0; s < 2; ++s) {
      fq[s].u[0] = *(const uint4*)(qrow + s * 32 + half * 8);
      fq[s].u[1] = *(const uint4*)(qrow + s * 32 + 16 + half * 8);
    }
  }

  float mrow[8], lrow[8];
  for (int vv = 0; vv < 8; ++vv) { mrow[vv] = -3.0e38f; lrow[vv] = 0.0f; }
  v8f accO[4];  // 16 rows x 64 (HD) output accumulator
  for (int t = 0; t < 4; ++t) accO[t] = z8;

  const int jend = qb + 16;  // causal: keys <= qb+15 (wave-local bound)
  for (int jb = 0; jb < jend; jb += 32) {
    // ---- scores = Q Kb^T : two 16x16 tiles (keys jb..+15, jb+16..+31) ----
    AccF sc[2];
    for (int t = 0; t < 2; ++t) {
      const u16* krow = Km + (size_t)(b * S_ + jb + t * 16 + ln) * D_ + h * HD_;
      v8f acc = z8;
      for (int s = 0; s < 2; ++s) {
        FragB fk;
        fk.u[0] = *(const uint4*)(krow + s * 32 + half * 8);
        fk.u[1] = *(const uint4*)(krow + s * 32 + 16 + half * 8);
        acc = wmma_bf16(fq[s].v, fk.v, acc);
      }
      sc[t].v = acc;
    }

    // ---- scale + causal mask + row max (16-lane shuffle reduce) ----
    float mx[8];
    for (int vv = 0; vv < 8; ++vv) {
      int qi = qb + half * 8 + vv;
      float a0 = sc[0].f[vv] * scale;
      float a1 = sc[1].f[vv] * scale;
      if (jb + ln > qi)      a0 = -1.0e30f;
      if (jb + 16 + ln > qi) a1 = -1.0e30f;
      sc[0].f[vv] = a0; sc[1].f[vv] = a1;
      float m = fmaxf(a0, a1);
      for (int off = 8; off >= 1; off >>= 1)
        m = fmaxf(m, __shfl_xor(m, off, 16));
      mx[vv] = m;
    }

    // ---- online softmax update ----
    float rs[8];
    for (int vv = 0; vv < 8; ++vv) {
      float mn  = fmaxf(mrow[vv], mx[vv]);
      float cor = __expf(mrow[vv] - mn);
      float p0  = __expf(sc[0].f[vv] - mn);
      float p1  = __expf(sc[1].f[vv] - mn);
      sc[0].f[vv] = p0; sc[1].f[vv] = p1;
      float ps = p0 + p1;
      for (int off = 8; off >= 1; off >>= 1)
        ps += __shfl_xor(ps, off, 16);
      lrow[vv] = lrow[vv] * cor + ps;
      mrow[vv] = mn;
      rs[vv]   = cor;
    }
    for (int t = 0; t < 4; ++t) {
      AccF o; o.v = accO[t];
      for (int vv = 0; vv < 8; ++vv) o.f[vv] *= rs[vv];
      accO[t] = o.v;
    }

    // ---- P (C-layout) -> LDS -> A-fragment layout ----
    for (int t = 0; t < 2; ++t)
      for (int vv = 0; vv < 8; ++vv)
        sP[w][(half * 8 + vv) * 32 + t * 16 + ln] = f2bf(sc[t].f[vv]);
    FragB fp;
    fp.u[0] = *(const uint4*)(&sP[w][ln * 32 + half * 8]);
    fp.u[1] = *(const uint4*)(&sP[w][ln * 32 + 16 + half * 8]);

    // ---- O += P @ V : four 16-wide HD tiles, K=32 keys ----
    for (int t = 0; t < 4; ++t) {
      const u16* vrow = Vt + ((size_t)bh * HD_ + t * 16 + ln) * S_ + jb;
      FragB fv;
      fv.u[0] = *(const uint4*)(vrow + half * 8);
      fv.u[1] = *(const uint4*)(vrow + 16 + half * 8);
      accO[t] = wmma_bf16(fp.v, fv.v, accO[t]);
    }
  }

  // ---- normalize + store Y (merge heads) ----
  float inv[8];
  for (int vv = 0; vv < 8; ++vv) inv[vv] = 1.0f / lrow[vv];
  for (int t = 0; t < 4; ++t) {
    AccF o; o.v = accO[t];
    for (int vv = 0; vv < 8; ++vv) {
      size_t row = (size_t)(b * S_ + qb + half * 8 + vv);
      Y[row * D_ + h * HD_ + t * 16 + ln] = f2bf(o.f[vv] * inv[vv]);
    }
  }
}

// ---------------- LayerNorm (block per row, D=1024, 256 thr x 4) ----------------
template <bool OUTF>
__global__ __launch_bounds__(256)
void k_ln(const float* __restrict__ x, const float* __restrict__ g,
          const float* __restrict__ be, u16* __restrict__ outb,
          float* __restrict__ outf) {
  __shared__ float rs1[256], rs2[256];
  const int tid = threadIdx.x;
  const float* xr = x + (size_t)blockIdx.x * D_;
  float4 xv = *(const float4*)(xr + tid * 4);
  float s  = xv.x + xv.y + xv.z + xv.w;
  float s2 = xv.x * xv.x + xv.y * xv.y + xv.z * xv.z + xv.w * xv.w;
  rs1[tid] = s; rs2[tid] = s2;
  __syncthreads();
  for (int off = 128; off > 0; off >>= 1) {
    if (tid < off) { rs1[tid] += rs1[tid + off]; rs2[tid] += rs2[tid + off]; }
    __syncthreads();
  }
  const float mu  = rs1[0] * (1.0f / D_);
  const float var = rs2[0] * (1.0f / D_) - mu * mu;
  const float r   = rsqrtf(var + 1e-5f);
  float* of = OUTF ? (outf + (size_t)blockIdx.x * D_) : nullptr;
  u16*   ob = OUTF ? nullptr : (outb + (size_t)blockIdx.x * D_);
  float vals[4] = {xv.x, xv.y, xv.z, xv.w};
  for (int j = 0; j < 4; ++j) {
    int col = tid * 4 + j;
    float y = (vals[j] - mu) * r * g[col] + be[col];
    if (OUTF) of[col] = y; else ob[col] = f2bf(y);
  }
}

// ---------------- host orchestration ----------------
extern "C" void kernel_launch(void* const* d_in, const int* in_sizes, int n_in,
                              void* d_out, int out_size, void* d_ws, size_t ws_size,
                              hipStream_t stream) {
  (void)in_sizes; (void)n_in; (void)out_size; (void)ws_size;
  const float* x    = (const float*)d_in[0];
  const float* Wq   = (const float*)d_in[1];  const float* bq = (const float*)d_in[2];
  const float* Wk   = (const float*)d_in[3];  const float* bk = (const float*)d_in[4];
  const float* Wv   = (const float*)d_in[5];  const float* bv = (const float*)d_in[6];
  const float* Wp   = (const float*)d_in[7];  const float* bp = (const float*)d_in[8];
  const float* g0   = (const float*)d_in[9];  const float* be0 = (const float*)d_in[10];
  const float* W1   = (const float*)d_in[11]; const float* b1 = (const float*)d_in[12];
  const float* W2   = (const float*)d_in[13]; const float* b2 = (const float*)d_in[14];
  const float* g1   = (const float*)d_in[15]; const float* be1 = (const float*)d_in[16];
  float* out = (float*)d_out;

  // workspace carve-out
  char* p = (char*)d_ws;
  auto take = [&](size_t bytes) { char* q = p; p += (bytes + 255) & ~(size_t)255; return q; };
  u16* xb  = (u16*)take((size_t)NT_ * D_ * 2);
  u16* WqT = (u16*)take((size_t)D_ * D_ * 2);
  u16* WkT = (u16*)take((size_t)D_ * D_ * 2);
  u16* WvT = (u16*)take((size_t)D_ * D_ * 2);
  u16* WpT = (u16*)take((size_t)D_ * D_ * 2);
  u16* W1T = (u16*)take((size_t)D_ * DFF_ * 2);
  u16* W2T = (u16*)take((size_t)DFF_ * D_ * 2);
  u16* Qb  = (u16*)take((size_t)NT_ * D_ * 2);
  u16* Kb  = (u16*)take((size_t)NT_ * D_ * 2);
  u16* Vb  = (u16*)take((size_t)NT_ * D_ * 2);
  u16* Vt  = (u16*)take((size_t)NT_ * D_ * 2);
  u16* Yb  = (u16*)take((size_t)NT_ * D_ * 2);
  float* x1 = (float*)take((size_t)NT_ * D_ * 4);
  u16* hb  = (u16*)take((size_t)NT_ * D_ * 2);
  u16* H1  = (u16*)take((size_t)NT_ * DFF_ * 2);
  float* x2 = (float*)take((size_t)NT_ * D_ * 4);

  const int nXD = NT_ * D_;
  const int nDD = D_ * D_;
  const int nDF = D_ * DFF_;

  // 1) casts / weight transposes
  k_cast_bf16<<<(nXD + 255) / 256, 256, 0, stream>>>(x, xb, nXD);
  k_transpose_w<<<(nDD + 255) / 256, 256, 0, stream>>>(Wq, WqT, D_, D_);
  k_transpose_w<<<(nDD + 255) / 256, 256, 0, stream>>>(Wk, WkT, D_, D_);
  k_transpose_w<<<(nDD + 255) / 256, 256, 0, stream>>>(Wv, WvT, D_, D_);
  k_transpose_w<<<(nDD + 255) / 256, 256, 0, stream>>>(Wp, WpT, D_, D_);
  k_transpose_w<<<(nDF + 255) / 256, 256, 0, stream>>>(W1, W1T, D_, DFF_);
  k_transpose_w<<<(nDF + 255) / 256, 256, 0, stream>>>(W2, W2T, DFF_, D_);

  // 2) QKV projections (bf16 out)
  dim3 gQ(D_ / 128, NT_ / 128);
  k_gemm<EP_BF16><<<gQ, 256, 0, stream>>>(xb, WqT, bq, NT_, D_, D_, nullptr, nullptr, Qb);
  k_gemm<EP_BF16><<<gQ, 256, 0, stream>>>(xb, WkT, bk, NT_, D_, D_, nullptr, nullptr, Kb);
  k_gemm<EP_BF16><<<gQ, 256, 0, stream>>>(xb, WvT, bv, NT_, D_, D_, nullptr, nullptr, Vb);

  // 3) V -> V^T per head
  k_transpose_v<<<(nXD + 255) / 256, 256, 0, stream>>>(Vb, Vt);

  // 4) causal flash attention -> Y bf16
  dim3 gF(S_ / 64, B_ * H_);
  k_flash<<<gF, 128, 0, stream>>>(Qb, Kb, Vt, Yb);

  // 5) output projection + residual: x1 = x + (Y Wp + bp)
  k_gemm<EP_RES><<<gQ, 256, 0, stream>>>(Yb, WpT, bp, NT_, D_, D_, x, x1, nullptr);

  // 6) LN0 -> bf16
  k_ln<false><<<NT_, 256, 0, stream>>>(x1, g0, be0, hb, nullptr);

  // 7) MLP up + GELU -> bf16
  dim3 g1d(DFF_ / 128, NT_ / 128);
  k_gemm<EP_GELU><<<g1d, 256, 0, stream>>>(hb, W1T, b1, NT_, DFF_, D_, nullptr, nullptr, H1);

  // 8) MLP down + residual: x2 = x1 + (H1 W2 + b2)
  k_gemm<EP_RES><<<gQ, 256, 0, stream>>>(H1, W2T, b2, NT_, D_, DFF_, x1, x2, nullptr);

  // 9) final LN -> fp32 out
  k_ln<true><<<NT_, 256, 0, stream>>>(x2, g1, be1, nullptr, out);
}
